// ExpertPool_4011499454968
// MI455X (gfx1250) — matrix operand order repaired and checked by reading, not compiled, not run on hardware
//
#include <hip/hip_runtime.h>
#include <hip/hip_bf16.h>
#include <stdint.h>

// ---------------------------------------------------------------------------
// MoE expert FFN (top-k routing) for MI455X / gfx1250.
//   out[t] = sum_k ( gelu(x[t] @ w1[e_k] + b1[e_k]) @ w2[e_k] + b2[e_k] )
// x/w1/w2 pre-converted to bf16 in workspace (~136MB; bf16 weights fit the
// 192MB L2).  All WMMA operands come from LDS; both weight streams are staged
// by double-buffered async DMA (ASYNCcnt).  Staging uses per-thread contiguous
// runs so each stage is one base address + literal instruction offsets
// (async-to-LDS applies INST_OFFSET to both the LDS and global address).
// ---------------------------------------------------------------------------

typedef __attribute__((ext_vector_type(16))) __bf16          v16bf;
typedef __attribute__((ext_vector_type(8)))  float           v8f;
typedef __attribute__((ext_vector_type(8)))  unsigned short  u16x8;

// Problem constants (from the reference).
constexpr int E = 8, D = 1024, F = 4096;
constexpr int Bb = 2, S = 2048, K = 2;
constexpr int T         = Bb * S;        // 4096 tokens
constexpr int CAP       = T * K;         // worst-case rows per expert (8192)
constexpr int MTILE     = 32;            // tokens per block (2 WMMA sub-tiles)
constexpr int MAX_TILES = CAP / MTILE;   // 256
constexpr int FC        = 128;           // F chunk (8 waves x 16 cols)
constexpr int KSTAGE    = 128;           // K rows of w1 staged in LDS per step

// LDS row pitches.  Staging runs must not cross padding, so pads sit at row
// ends; 8-halfword pad still breaks the 256B bank period for fragment reads.
constexpr int XS_LD  = D  + 8;           // 1032 halfwords
constexpr int WS1_LD = FC + 8;           // 136 halfwords
constexpr int WS2_LD = D  + 8;           // 1032 halfwords
constexpr int HS_LD  = FC + 8;           // 136 halfwords

// Workspace layout (bytes).
constexpr size_t WS_COUNTS = 0;                               // E x u32
constexpr size_t WS_LISTS  = 256;                             // E*CAP x i32
constexpr size_t WS_W1BF   = 512 * 1024;                      // E*D*F bf16
constexpr size_t WS_W2BF   = WS_W1BF + (size_t)E * D * F * 2; // E*F*D bf16
constexpr size_t WS_XBF    = WS_W2BF + (size_t)E * F * D * 2; // T*D bf16

__device__ __forceinline__ unsigned short f2bf(float f) {
  unsigned u = __float_as_uint(f);
  u += 0x7FFFu + ((u >> 16) & 1u);       // round-to-nearest-even
  return (unsigned short)(u >> 16);
}

__device__ __forceinline__ v16bf concat8(u16x8 a, u16x8 b) {
  return __builtin_bit_cast(
      v16bf, __builtin_shufflevector(a, b, 0, 1, 2, 3, 4, 5, 6, 7,
                                     8, 9, 10, 11, 12, 13, 14, 15));
}

// 16-bit A fragment (16x32, lane -> M row): lanes 0-15 hold K {0-7,16-23},
// lanes 16-31 hold K {8-15,24-31}.  rowp points at (row m, k-step base).
__device__ __forceinline__ v16bf load_a_frag(const unsigned short* rowp, int ah) {
  u16x8 a0 = *(const u16x8*)(rowp + ah * 8);
  u16x8 a1 = *(const u16x8*)(rowp + ah * 8 + 16);
  return concat8(a0, a1);
}

__device__ __forceinline__ void wmma_bf16(v16bf A, v16bf B, v8f& C) {
  C = __builtin_amdgcn_wmma_f32_16x16x32_bf16(false, A, false, B, (short)0, C,
                                              false, false);
}

__device__ __forceinline__ unsigned lds_addr(const void* p) {
  return (unsigned)(uintptr_t)p;
}

// Async DMA global->LDS, INST_OFFSET applies to both LDS and global address.
#define ASYNC_B128(lds, gaddr, OFF)                                       \
  asm volatile("global_load_async_to_lds_b128 %0, %1, off offset:" #OFF   \
               :: "v"(lds), "v"(gaddr) : "memory")

// 8 x 16B = 128B contiguous run (w1 staging).
__device__ __forceinline__ void stage128_async(unsigned lds,
                                               const unsigned short* g) {
  unsigned long long a = (unsigned long long)(uintptr_t)g;
  ASYNC_B128(lds, a, 0);   ASYNC_B128(lds, a, 16);
  ASYNC_B128(lds, a, 32);  ASYNC_B128(lds, a, 48);
  ASYNC_B128(lds, a, 64);  ASYNC_B128(lds, a, 80);
  ASYNC_B128(lds, a, 96);  ASYNC_B128(lds, a, 112);
}

// 16 x 16B = 256B contiguous run (w2 / x staging).
__device__ __forceinline__ void stage256_async(unsigned lds,
                                               const unsigned short* g) {
  unsigned long long a = (unsigned long long)(uintptr_t)g;
  ASYNC_B128(lds, a, 0);   ASYNC_B128(lds, a, 16);
  ASYNC_B128(lds, a, 32);  ASYNC_B128(lds, a, 48);
  ASYNC_B128(lds, a, 64);  ASYNC_B128(lds, a, 80);
  ASYNC_B128(lds, a, 96);  ASYNC_B128(lds, a, 112);
  ASYNC_B128(lds, a, 128); ASYNC_B128(lds, a, 144);
  ASYNC_B128(lds, a, 160); ASYNC_B128(lds, a, 176);
  ASYNC_B128(lds, a, 192); ASYNC_B128(lds, a, 208);
  ASYNC_B128(lds, a, 224); ASYNC_B128(lds, a, 240);
}

// ---------------------------------------------------------------------------
__global__ void zero_kernel(float* __restrict__ out, unsigned* __restrict__ counts) {
  int i = blockIdx.x * blockDim.x + threadIdx.x;   // grid covers T*D/4 exactly
  float4 z = {0.f, 0.f, 0.f, 0.f};
  ((float4*)out)[i] = z;
  if (blockIdx.x == 0 && threadIdx.x < E) counts[threadIdx.x] = 0u;
}

__global__ void route_kernel(const int* __restrict__ idx,
                             unsigned* __restrict__ counts,
                             int* __restrict__ lists) {
  int i = blockIdx.x * blockDim.x + threadIdx.x;   // 0 .. T*K-1
  if (i >= T * K) return;
  int t = i / K;
  int e = idx[i];
  unsigned p = atomicAdd(&counts[e], 1u);
  lists[e * CAP + p] = t;
}

// f32 -> bf16 pre-conversion, 8 elements/thread, fully coalesced.
__global__ void convert_bf16_kernel(const float* __restrict__ src,
                                    unsigned short* __restrict__ dst) {
  size_t i = ((size_t)blockIdx.x * blockDim.x + threadIdx.x) * 8;
  float4 a = *(const float4*)(src + i);
  float4 b = *(const float4*)(src + i + 4);
  u16x8 o;
  o[0] = f2bf(a.x); o[1] = f2bf(a.y); o[2] = f2bf(a.z); o[3] = f2bf(a.w);
  o[4] = f2bf(b.x); o[5] = f2bf(b.y); o[6] = f2bf(b.z); o[7] = f2bf(b.w);
  *(u16x8*)(dst + i) = o;
}

// ---------------------------------------------------------------------------
__global__ __launch_bounds__(256) void moe_ffn_kernel(
    const unsigned short* __restrict__ xbf,
    const unsigned short* __restrict__ w1bf,
    const float* __restrict__ b1,
    const unsigned short* __restrict__ w2bf,
    const float* __restrict__ b2,
    const unsigned* __restrict__ counts,
    const int* __restrict__ lists, float* __restrict__ out) {
  const int e    = blockIdx.x / MAX_TILES;
  const int tile = blockIdx.x % MAX_TILES;
  const int nrows_total = (int)counts[e];
  const int row0 = tile * MTILE;
  if (row0 >= nrows_total) return;                 // empty tile -> cheap exit
  const int rows = min(MTILE, nrows_total - row0);

  __shared__ unsigned short xs[MTILE * XS_LD];         // x tile bf16    (~66 KB)
  __shared__ unsigned short ws1[2][KSTAGE * WS1_LD];   // w1 ping-pong   (~70 KB)
  __shared__ unsigned short ws2[2][32 * WS2_LD];       // w2 ping-pong  (~132 KB)
  __shared__ unsigned short hs[MTILE * HS_LD];         // gelu(h) bf16    (~9 KB)
  __shared__ int toks[MTILE];

  const int tid  = threadIdx.x;
  const int wave = tid >> 5;
  const int lane = tid & 31;
  const int am   = lane & 15;            // A fragment M row / C-D N column
  const int ah   = lane >> 4;            // A fragment K-half select
  const int nbase_wave = wave * 16;      // this wave's F columns within chunk
  const int dbase_wave = wave * 128;     // this wave's D output columns

  if (tid < MTILE) toks[tid] = lists[e * CAP + row0 + min(tid, rows - 1)];
  __syncthreads();

  // ---- per-thread staging bases (computed once; stages add uniform offs) --
  const unsigned short* w1e = w1bf + (size_t)e * D * F;
  const unsigned short* w2e = w2bf + (size_t)e * F * D;
  // w1 tile 128x128: thread t -> row t>>1, 128B run at col (t&1)*64
  const unsigned short* w1_src0 = w1e + (size_t)(tid >> 1) * F + (tid & 1) * 64;
  const unsigned  ws1_dst0 = lds_addr(&ws1[0][(tid >> 1) * WS1_LD + (tid & 1) * 64]);
  const unsigned  ws1_dst1 = lds_addr(&ws1[1][(tid >> 1) * WS1_LD + (tid & 1) * 64]);
  // w2 block 32x1024: thread t -> row t>>3, 256B run at col (t&7)*128
  const unsigned short* w2_src0 = w2e + (size_t)(tid >> 3) * D + (tid & 7) * 128;
  const unsigned  ws2_dst0 = lds_addr(&ws2[0][(tid >> 3) * WS2_LD + (tid & 7) * 128]);
  const unsigned  ws2_dst1 = lds_addr(&ws2[1][(tid >> 3) * WS2_LD + (tid & 7) * 128]);

  // ---- async gather of 32 token rows (bf16) into LDS ----
  stage256_async(lds_addr(&xs[(tid >> 3) * XS_LD + (tid & 7) * 128]),
                 xbf + (size_t)toks[tid >> 3] * D + (tid & 7) * 128);

  v8f acc[2][8] = {};                    // 32x128 f32 output tile per wave

  for (int fc = 0; fc < F; fc += FC) {
    v8f hacc0 = {}, hacc1 = {};
    // ============ GEMM1: h = x @ w1[:, fc:fc+128], async-staged ============
    stage128_async(ws1_dst0, w1_src0 + fc);
    asm volatile("s_wait_asynccnt 0" ::: "memory");   // covers x DMA at fc==0
    __syncthreads();
    for (int kb = 0; kb < D; kb += KSTAGE) {
      const int cur = (kb / KSTAGE) & 1;
      if (kb + KSTAGE < D)               // prefetch next tile via async DMA
        stage128_async(cur ? ws1_dst0 : ws1_dst1,
                       w1_src0 + (size_t)(kb + KSTAGE) * F + fc);
#pragma unroll
      for (int ks = 0; ks < KSTAGE; ks += 32) {
        v16bf A0 = load_a_frag(&xs[am * XS_LD + kb + ks], ah);
        v16bf A1 = load_a_frag(&xs[(am + 16) * XS_LD + kb + ks], ah);
        // B fragment (32x16, lane -> N col, K in A-fragment order)
        const unsigned short* bp = &ws1[cur][(ks + ah * 8) * WS1_LD + nbase_wave + am];
        u16x8 b0, b1v;
#pragma unroll
        for (int i = 0; i < 8; ++i) { b0[i] = bp[i * WS1_LD]; b1v[i] = bp[(i + 16) * WS1_LD]; }
        v16bf Bf = concat8(b0, b1v);
        wmma_bf16(A0, Bf, hacc0);
        wmma_bf16(A1, Bf, hacc1);
      }
      asm volatile("s_wait_asynccnt 0" ::: "memory");
      __syncthreads();
    }
    // ==== GELU(h + b1) -> LDS bf16, overlapped with w2 ks=0 DMA stage ====
    stage256_async(ws2_dst0, w2_src0 + (size_t)fc * D);
    {
      float b1v = b1[e * F + fc + nbase_wave + am];
      int mbase = ah * 8;
#pragma unroll
      for (int r = 0; r < 8; ++r) {
        float h0 = hacc0[r] + b1v;
        float h1 = hacc1[r] + b1v;
        h0 = 0.5f * h0 * (1.0f + erff(h0 * 0.70710678118f));
        h1 = 0.5f * h1 * (1.0f + erff(h1 * 0.70710678118f));
        hs[(mbase + r) * HS_LD + nbase_wave + am] = f2bf(h0);
        hs[(mbase + r + 16) * HS_LD + nbase_wave + am] = f2bf(h1);
      }
    }
    asm volatile("s_wait_asynccnt 0" ::: "memory");
    __syncthreads();                     // hs ready AND ws2[0] ready
    // ====== GEMM2: acc += h @ w2[fc:fc+128, :], LDS double-buffered ======
    for (int kblk = 0; kblk < 4; ++kblk) {
      const int cur = kblk & 1;
      if (kblk < 3)                      // prefetch next 32x1024 block via DMA
        stage256_async(cur ? ws2_dst0 : ws2_dst1,
                       w2_src0 + (size_t)(fc + (kblk + 1) * 32) * D);
      v16bf A0 = load_a_frag(&hs[am * HS_LD + kblk * 32], ah);
      v16bf A1 = load_a_frag(&hs[(am + 16) * HS_LD + kblk * 32], ah);
#pragma unroll
      for (int nt = 0; nt < 8; ++nt) {
        // B fragment (32x16, lane -> N col, K in A-fragment order) from LDS
        const unsigned short* bp =
            &ws2[cur][(ah * 8) * WS2_LD + dbase_wave + nt * 16 + am];
        u16x8 b0, b1v;
#pragma unroll
        for (int i = 0; i < 8; ++i) { b0[i] = bp[i * WS2_LD]; b1v[i] = bp[(i + 16) * WS2_LD]; }
        v16bf Bf = concat8(b0, b1v);
        wmma_bf16(A0, Bf, acc[0][nt]);
        wmma_bf16(A1, Bf, acc[1][nt]);
      }
      asm volatile("s_wait_asynccnt 0" ::: "memory");
      __syncthreads();
    }
  }

  // ---- epilogue: + b2, scatter-add valid rows into out ----
  {
    int mbase = ah * 8;
#pragma unroll
    for (int nt = 0; nt < 8; ++nt) {
      int d = dbase_wave + nt * 16 + am;
      float b2v = b2[e * D + d];
#pragma unroll
      for (int r = 0; r < 8; ++r) {
        int m0 = mbase + r;
        if (m0 < rows)
          atomicAdd(&out[(size_t)toks[m0] * D + d], acc[0][nt][r] + b2v);
        int m1 = mbase + r + 16;
        if (m1 < rows)
          atomicAdd(&out[(size_t)toks[m1] * D + d], acc[1][nt][r] + b2v);
      }
    }
  }
}

// ---------------------------------------------------------------------------
extern "C" void kernel_launch(void* const* d_in, const int* in_sizes, int n_in,
                              void* d_out, int out_size, void* d_ws, size_t ws_size,
                              hipStream_t stream) {
  (void)in_sizes; (void)n_in; (void)out_size; (void)ws_size;
  const float* x  = (const float*)d_in[0];
  const int*   id = (const int*)d_in[1];
  const float* w1 = (const float*)d_in[2];
  const float* b1 = (const float*)d_in[3];
  const float* w2 = (const float*)d_in[4];
  const float* b2 = (const float*)d_in[5];
  float* out = (float*)d_out;

  unsigned*       counts = (unsigned*)((char*)d_ws + WS_COUNTS);
  int*            lists  = (int*)((char*)d_ws + WS_LISTS);
  unsigned short* w1bf   = (unsigned short*)((char*)d_ws + WS_W1BF);
  unsigned short* w2bf   = (unsigned short*)((char*)d_ws + WS_W2BF);
  unsigned short* xbf    = (unsigned short*)((char*)d_ws + WS_XBF);

  constexpr size_t WELEMS = (size_t)E * D * F;       // 32M per weight tensor
  constexpr size_t XELEMS = (size_t)T * D;           // 4M

  zero_kernel<<<(T * D / 4) / 256, 256, 0, stream>>>(out, counts);
  route_kernel<<<(T * K + 255) / 256, 256, 0, stream>>>(id, counts, lists);
  convert_bf16_kernel<<<WELEMS / (8 * 256), 256, 0, stream>>>(w1, w1bf);
  convert_bf16_kernel<<<WELEMS / (8 * 256), 256, 0, stream>>>(w2, w2bf);
  convert_bf16_kernel<<<XELEMS / (8 * 256), 256, 0, stream>>>(x, xbf);
  moe_ffn_kernel<<<E * MAX_TILES, 256, 0, stream>>>(xbf, w1bf, b1, w2bf, b2,
                                                    counts, lists, out);
}
